// SparseConv3D_32822140076334
// MI455X (gfx1250) — matrix operand order repaired and checked by reading, not compiled, not run on hardware
//
#include <hip/hip_runtime.h>

typedef __attribute__((ext_vector_type(16))) _Float16 v16h;
typedef __attribute__((ext_vector_type(8)))  float    v8f;

#define SDIM 128
#define GRID_KEYS (SDIM * SDIM * SDIM)   // 2,097,152 possible output coords
#define SCAN_BLOCK 256
#define SCAN_ITEMS 16
#define SCAN_TILE  (SCAN_BLOCK * SCAN_ITEMS)   // 4096
#define SCAN_NBLK  (GRID_KEYS / SCAN_TILE)     // 512 (exact)

// ---------------------------------------------------------------------------
// Output init: uniq region (NK*3 ints) <- 128 (matches jnp.unique fill_value),
// feature region (NK*64 floats) <- 0.0f (bits 0).
// ---------------------------------------------------------------------------
__global__ void k_init_out(unsigned* __restrict__ out, long long total, long long uniqInts) {
    long long i = (long long)blockIdx.x * blockDim.x + threadIdx.x;
    if (i < total) out[i] = (i < uniqInts) ? 128u : 0u;
}

__global__ void k_zero(unsigned* __restrict__ p, int n) {
    int i = blockIdx.x * blockDim.x + threadIdx.x;
    if (i < n) p[i] = 0u;
}

// ---------------------------------------------------------------------------
// Expand each (point, tap) entry -> output coordinate key; mark dense grid.
// key = (x<<14)|(y<<7)|z is monotone in lexicographic (x,y,z) order.
// ---------------------------------------------------------------------------
__global__ void k_expand(const int* __restrict__ idx, int* __restrict__ keys,
                         unsigned* __restrict__ mark, int NK) {
    int i = blockIdx.x * blockDim.x + threadIdx.x;
    if (i >= NK) return;
    int n = i / 27;
    int t = i - n * 27;
    int x = idx[3 * n + 0] - (t / 9 - 1);
    int y = idx[3 * n + 1] - ((t / 3) % 3 - 1);
    int z = idx[3 * n + 2] - (t % 3 - 1);
    int key = -1;
    if ((unsigned)x < 128u && (unsigned)y < 128u && (unsigned)z < 128u) {
        key = (x << 14) | (y << 7) | z;
        mark[key] = 1u;   // benign race: all writers store 1 (deterministic)
    }
    keys[i] = key;
}

// ---------------------------------------------------------------------------
// Exclusive prefix scan over the 2.1M-entry mark grid (3 kernels).
// rank[k] = number of occupied keys < k  == row index in sorted unique output.
// ---------------------------------------------------------------------------
__global__ void k_scan_reduce(const unsigned* __restrict__ mark, unsigned* __restrict__ bsum) {
    __shared__ unsigned sh[SCAN_BLOCK];
    int t = threadIdx.x;
    long long base = (long long)blockIdx.x * SCAN_TILE + (long long)t * SCAN_ITEMS;
    unsigned s = 0;
#pragma unroll
    for (int j = 0; j < SCAN_ITEMS; ++j) s += mark[base + j];
    sh[t] = s;
    __syncthreads();
    for (int off = SCAN_BLOCK / 2; off > 0; off >>= 1) {
        if (t < off) sh[t] += sh[t + off];
        __syncthreads();
    }
    if (t == 0) bsum[blockIdx.x] = sh[0];
}

__global__ void k_scan_sums(unsigned* __restrict__ bsum) {  // 1 block, SCAN_NBLK threads
    __shared__ unsigned sh[SCAN_NBLK];
    int t = threadIdx.x;
    unsigned v = bsum[t];
    sh[t] = v;
    __syncthreads();
    for (int off = 1; off < SCAN_NBLK; off <<= 1) {
        unsigned u = (t >= off) ? sh[t - off] : 0u;
        __syncthreads();
        sh[t] += u;
        __syncthreads();
    }
    bsum[t] = sh[t] - v;  // exclusive
}

__global__ void k_scan_apply(const unsigned* __restrict__ mark, const unsigned* __restrict__ bsum,
                             unsigned* __restrict__ rank) {
    __shared__ unsigned sh[SCAN_BLOCK];
    int t = threadIdx.x;
    long long base = (long long)blockIdx.x * SCAN_TILE + (long long)t * SCAN_ITEMS;
    unsigned v[SCAN_ITEMS];
    unsigned s = 0;
#pragma unroll
    for (int j = 0; j < SCAN_ITEMS; ++j) { v[j] = mark[base + j]; s += v[j]; }
    sh[t] = s;
    __syncthreads();
    for (int off = 1; off < SCAN_BLOCK; off <<= 1) {
        unsigned u = (t >= off) ? sh[t - off] : 0u;
        __syncthreads();
        sh[t] += u;
        __syncthreads();
    }
    unsigned run = bsum[blockIdx.x] + (sh[t] - s);
#pragma unroll
    for (int j = 0; j < SCAN_ITEMS; ++j) { rank[base + j] = run; run += v[j]; }
}

// Emit the sorted unique coordinates.
__global__ void k_uniq(const unsigned* __restrict__ mark, const unsigned* __restrict__ rank,
                       int* __restrict__ uniqOut) {
    int k = blockIdx.x * blockDim.x + threadIdx.x;
    if (k >= GRID_KEYS) return;
    if (mark[k]) {
        unsigned r = rank[k];
        uniqOut[3 * r + 0] = k >> 14;
        uniqOut[3 * r + 1] = (k >> 7) & 127;
        uniqOut[3 * r + 2] = k & 127;
    }
}

// Per-entry output-row lookup (inv of jnp.unique); -1 for invalid entries.
__global__ void k_pos(const int* __restrict__ keys, const unsigned* __restrict__ rank,
                      int* __restrict__ pos, int NK) {
    int i = blockIdx.x * blockDim.x + threadIdx.x;
    if (i >= NK) return;
    int k = keys[i];
    pos[i] = (k >= 0) ? (int)rank[k] : -1;
}

// ---------------------------------------------------------------------------
// Gather-matmul via WMMA: one wave handles 16 entries x 64 channels.
//   A (16x32 f16): row m = one-hot(tap[m]) * feat[m]   (taps 0..26, K padded to 32)
//   B (32x16 f16): K-th row = kernel[K, cb .. cb+15]
//   D = A x B  => D[m, n] = feat[m] * kernel[tap[m], cb + n]
// Then scatter-add D into the output with global f32 atomics using the
// documented C/D layout: VGPR v, lane l -> (M = v + 8*(l>=16), N = l%16).
// ---------------------------------------------------------------------------
__global__ void k_wmma_scatter(const float* __restrict__ feats, const float* __restrict__ kern,
                               const int* __restrict__ pos, float* __restrict__ outF,
                               int NK, int nWaves) {
    int wave = blockIdx.x * (blockDim.x >> 5) + (threadIdx.x >> 5);
    if (wave >= nWaves) return;           // wave-uniform: EXEC all-ones below
    int lane = threadIdx.x & 31;
    int hi   = lane >> 4;                 // which half of the K/M striping
    int m    = lane & 15;
    int base = wave * 16;

    // ---- Build A (one-hot taps scaled by feature) ----
    int entry = base + m;
    int tap = 0;
    float f = 0.0f;
    if (entry < NK) { tap = entry % 27; f = feats[entry / 27]; }
    _Float16 fh = (_Float16)f;
    // 16-bit A 16x32 layout (ISA 7.12.2): lane-lo holds K in {0..7,16..23},
    // lane-hi holds K in {8..15,24..31}; element e maps linearly within each half.
    int eA = -1;
    if (!hi) {
        if (tap < 8) eA = tap;
        else if (tap >= 16 && tap < 24) eA = tap - 8;
    } else {
        if (tap >= 8 && tap < 16) eA = tap - 8;
        else if (tap >= 24 && tap < 32) eA = tap - 16;
    }
    v16h a;
#pragma unroll
    for (int e = 0; e < 16; ++e) a[e] = (e == eA) ? fh : (_Float16)0.0f;

    // ---- Per-lane output rows for the scatter (M = v + 8*hi) ----
    int pr[8];
#pragma unroll
    for (int v = 0; v < 8; ++v) {
        int e2 = base + v + hi * 8;
        pr[v] = (e2 < NK) ? pos[e2] : -1;
    }

    // ---- 4 channel tiles of 16 ----
#pragma unroll
    for (int cbi = 0; cbi < 4; ++cbi) {
        int cb = cbi * 16;
        // B 32x16 layout: lanes 0-15 hold K in {0..7,16..23} (col N=lane), lanes 16-31 K in {8..15,24..31}.
        v16h b;
#pragma unroll
        for (int e = 0; e < 16; ++e) {
            int K = e + hi * 8 + ((e >= 8) ? 8 : 0);
            b[e] = (K < 27) ? (_Float16)kern[K * 64 + cb + m] : (_Float16)0.0f;
        }
        v8f c = {0.f, 0.f, 0.f, 0.f, 0.f, 0.f, 0.f, 0.f};
        c = __builtin_amdgcn_wmma_f32_16x16x32_f16(
                /*neg_a=*/false, a, /*neg_b=*/false, b,
                /*c_mod=*/(short)0, c, /*reuse_a=*/false, /*reuse_b=*/false);
#pragma unroll
        for (int v = 0; v < 8; ++v) {
            int p = pr[v];
            if (p >= 0)
                atomicAdd(&outF[(long long)p * 64 + cb + m], c[v]);
        }
    }
}

// ---------------------------------------------------------------------------
extern "C" void kernel_launch(void* const* d_in, const int* in_sizes, int n_in,
                              void* d_out, int out_size, void* d_ws, size_t ws_size,
                              hipStream_t stream) {
    (void)n_in; (void)out_size; (void)ws_size;
    const int*   indices = (const int*)d_in[0];    // [N,3] int32
    const float* feats   = (const float*)d_in[1];  // [N,1] f32
    const float* kern    = (const float*)d_in[2];  // [27,1,64] f32

    const int N  = in_sizes[0] / 3;
    const int NK = N * 27;

    // Workspace carve-out (256B aligned): ~27.6 MB total.
    char* ws = (char*)d_ws;
    size_t off = 0;
    auto carve = [&](size_t bytes) {
        void* p = ws + off;
        off = (off + bytes + 255) & ~(size_t)255;
        return p;
    };
    unsigned* mark = (unsigned*)carve((size_t)GRID_KEYS * 4);
    unsigned* rank = (unsigned*)carve((size_t)GRID_KEYS * 4);
    unsigned* bsum = (unsigned*)carve((size_t)SCAN_NBLK * 4);
    int*      keys = (int*)carve((size_t)NK * 4);
    int*      pos  = (int*)carve((size_t)NK * 4);

    int*   uniqOut = (int*)d_out;
    float* outF    = (float*)((char*)d_out + (size_t)NK * 3 * sizeof(int));

    const long long uniqInts = (long long)NK * 3;
    const long long total    = uniqInts + (long long)NK * 64;

    // 1) init output (fill coords = 128, feats = 0)
    {
        long long blocks = (total + 255) / 256;
        k_init_out<<<(int)blocks, 256, 0, stream>>>((unsigned*)d_out, total, uniqInts);
    }
    // 2) zero dense occupancy grid
    k_zero<<<(GRID_KEYS + 255) / 256, 256, 0, stream>>>(mark, GRID_KEYS);
    // 3) expand entries, mark grid
    k_expand<<<(NK + 255) / 256, 256, 0, stream>>>(indices, keys, mark, NK);
    // 4) exclusive scan over grid -> unique ranks (lexicographic order)
    k_scan_reduce<<<SCAN_NBLK, SCAN_BLOCK, 0, stream>>>(mark, bsum);
    k_scan_sums<<<1, SCAN_NBLK, 0, stream>>>(bsum);
    k_scan_apply<<<SCAN_NBLK, SCAN_BLOCK, 0, stream>>>(mark, bsum, rank);
    // 5) write sorted unique coordinates
    k_uniq<<<(GRID_KEYS + 255) / 256, 256, 0, stream>>>(mark, rank, uniqOut);
    // 6) per-entry output-row lookup
    k_pos<<<(NK + 255) / 256, 256, 0, stream>>>(keys, rank, pos, NK);
    // 7) WMMA gather-matmul + atomic scatter-add
    {
        int nWaves = (NK + 15) / 16;
        int wavesPerBlock = 256 / 32;  // 8 waves per block (wave32)
        int blocks = (nWaves + wavesPerBlock - 1) / wavesPerBlock;
        k_wmma_scatter<<<blocks, 256, 0, stream>>>(feats, kern, pos, outF, NK, nWaves);
    }
}
